// Attention_29935922053658
// MI455X (gfx1250) — compile-verified, matrix operand order checked by reading, not run on hardware
//
#include <hip/hip_runtime.h>
#include <hip/hip_bf16.h>

typedef __bf16 bf16_t;
typedef __attribute__((ext_vector_type(16))) __bf16 v16bf;
typedef __attribute__((ext_vector_type(8)))  __bf16 v8bf;
typedef __attribute__((ext_vector_type(8)))  float  v8f;

#define HEADS    8
#define DIM_HEAD 64
#define BATCH    8
#define FRAMES   16
#define NPATCH   196
#define DIM      512
#define SEQ      3137          /* 1 + 16*196 */
#define MROWS    25096         /* BATCH*SEQ  */
#define NQKV     1536
#define MTILES   1569          /* ceil(MROWS/16) */

#define NKPAD    224           /* 197 keys padded to 14 tiles of 16 */
#define KST      72            /* LDS row stride (elems) for K, 144B (16B aligned, bank-spread) */
#define VTST     232           /* LDS row stride for V^T (64 rows x 224 keys), 464B */
#define PST      232           /* LDS row stride for P */

static __device__ __forceinline__ v8f wmma_bf16(v16bf a, v16bf b, v8f c) {
  // D = A(16x32 bf16) * B(32x16 bf16) + C(16x16 f32)
  return __builtin_amdgcn_wmma_f32_16x16x32_bf16(false, a, false, b, (short)0, c,
                                                 false, false);
}

// Fragment gather: elements [0..7] at p[0..7], [8..15] at p[16..23].
// p must be 16B aligned -> two b128 loads (ds_load_b128 / global_load_b128).
static __device__ __forceinline__ v16bf ld_frag(const bf16_t* p) {
  v8bf lo = *(const v8bf*)p;
  v8bf hi = *(const v8bf*)(p + 16);
  return __builtin_shufflevector(lo, hi, 0, 1, 2, 3, 4, 5, 6, 7,
                                 8, 9, 10, 11, 12, 13, 14, 15);
}

// ---------------------------------------------------------------- converts
__global__ void convert_bf16_kernel(const float* __restrict__ src,
                                    bf16_t* __restrict__ dst, int n) {
  int i = blockIdx.x * blockDim.x + threadIdx.x;
  if (i < n) dst[i] = (bf16_t)src[i];
}

// dst[n * rows + k] = (bf16)src[k * cols + n]  (transpose K-major for B-fragments)
__global__ void convert_transpose_kernel(const float* __restrict__ src,
                                         bf16_t* __restrict__ dst,
                                         int rows, int cols) {
  int i = blockIdx.x * blockDim.x + threadIdx.x;
  if (i < rows * cols) {
    int n = i / rows;
    int k = i - n * rows;
    dst[i] = (bf16_t)src[(long)k * cols + n];
  }
}

// ---------------------------------------------------------------- QKV GEMM
// C[m, j] = xh[m, :] @ Wqkv[:, j]; scatter-store into q/k/v (B*h, SEQ, 64) bf16.
// wt is transposed: wt[j][k].
__global__ __launch_bounds__(256) void qkv_gemm_kernel(
    const bf16_t* __restrict__ xh, const bf16_t* __restrict__ wt,
    bf16_t* __restrict__ qh, bf16_t* __restrict__ kh, bf16_t* __restrict__ vh) {
  const int lane = threadIdx.x & 31;
  const int wave = threadIdx.x >> 5;
  const int mt = blockIdx.y * 8 + wave;
  if (mt >= MTILES) return;
  const int m0 = mt * 16;
  const int n0 = blockIdx.x * 16;
  const int am = lane & 15;              // A row / B col / C col for this lane
  const int kb = (lane >> 4) * 8;        // K sub-offset per 16-lane group
  const int grp = (lane >> 4) << 3;
  int arow = m0 + am; if (arow > MROWS - 1) arow = MROWS - 1;
  const bf16_t* xrow = xh + (long)arow * DIM;
  const bf16_t* wrow = wt + (long)(n0 + am) * DIM;

  v8f acc = {};
  for (int k0 = 0; k0 < DIM; k0 += 32) {
    __builtin_prefetch(xrow + k0 + 128, 0, 0);   // global_prefetch_b8
    v16bf a  = ld_frag(xrow + k0 + kb);
    v16bf bm = ld_frag(wrow + k0 + kb);
    acc = wmma_bf16(a, bm, acc);
  }

  const int col  = n0 + am;
  const int part = col >> 9;             // 0=q 1=k 2=v
  const int rem  = col & 511;
  const int head = rem >> 6;
  const int d    = rem & 63;
  bf16_t* dst = (part == 0) ? qh : ((part == 1) ? kh : vh);
  const float scale = (part == 0) ? 0.125f : 1.0f;   // dh^-0.5 folded into q
  const int b0 = m0 / SEQ;               // one divide per wave; rows cross SEQ
  const int s0 = m0 - b0 * SEQ;          // boundary at most once within a tile
#pragma unroll
  for (int r = 0; r < 8; ++r) {
    int m = m0 + r + grp;
    if (m < MROWS) {
      int s = s0 + r + grp;
      int b = b0;
      if (s >= SEQ) { s -= SEQ; ++b; }
      dst[(((long)(b * HEADS + head)) * SEQ + s) * DIM_HEAD + d] =
          (bf16_t)(acc[r] * scale);
    }
  }
}

// ---------------------------------------------------------------- CLS attention
// One block per (b*h): cls_q attends over all 3137 keys. Tiny; VALU + ds_add_f32.
__global__ __launch_bounds__(256) void cls_attn_kernel(
    const bf16_t* __restrict__ qh, const bf16_t* __restrict__ kh,
    const bf16_t* __restrict__ vh, bf16_t* __restrict__ attnout) {
  __shared__ float qs[DIM_HEAD];
  __shared__ float red[256];
  __shared__ float accsh[DIM_HEAD];
  const int bh = blockIdx.x;
  const int b = bh >> 3, h = bh & 7;
  const int tid = threadIdx.x;
  const long base = (long)bh * SEQ;
  if (tid < DIM_HEAD) { qs[tid] = (float)qh[base * DIM_HEAD + tid]; accsh[tid] = 0.f; }
  __syncthreads();

  float sv[13];
  int cnt = 0;
  float lmax = -1e30f;
  for (int j = tid; j < SEQ; j += 256) {
    const bf16_t* kp = kh + (base + j) * DIM_HEAD;
    float s = 0.f;
#pragma unroll
    for (int dd = 0; dd < DIM_HEAD; ++dd) s += qs[dd] * (float)kp[dd];
    sv[cnt++] = s;
    lmax = fmaxf(lmax, s);
  }
  red[tid] = lmax; __syncthreads();
  for (int off = 128; off > 0; off >>= 1) {
    if (tid < off) red[tid] = fmaxf(red[tid], red[tid + off]);
    __syncthreads();
  }
  const float gmax = red[0];
  __syncthreads();

  float pacc[DIM_HEAD];
#pragma unroll
  for (int dd = 0; dd < DIM_HEAD; ++dd) pacc[dd] = 0.f;
  float lsum = 0.f;
  cnt = 0;
  for (int j = tid; j < SEQ; j += 256) {
    float p = __expf(sv[cnt++] - gmax);
    lsum += p;
    const bf16_t* vp = vh + (base + j) * DIM_HEAD;
#pragma unroll
    for (int dd = 0; dd < DIM_HEAD; ++dd) pacc[dd] += p * (float)vp[dd];
  }
  red[tid] = lsum; __syncthreads();
  for (int off = 128; off > 0; off >>= 1) {
    if (tid < off) red[tid] += red[tid + off];
    __syncthreads();
  }
  const float gsum = red[0];
#pragma unroll
  for (int dd = 0; dd < DIM_HEAD; ++dd) atomicAdd(&accsh[dd], pacc[dd]);  // ds_add_f32
  __syncthreads();
  if (tid < DIM_HEAD)
    attnout[((long)b * SEQ) * DIM + h * DIM_HEAD + tid] = (bf16_t)(accsh[tid] / gsum);
}

// ---------------------------------------------------------------- frame attention
// One block per (f, b*h). keys = [cls] ++ frame tokens (197, pad to 224).
// K staged row-major (contraction d contiguous); V staged TRANSPOSED
// (Vt[d][key], contraction key contiguous) so every fragment is 2x b128 LDS loads.
__global__ __launch_bounds__(256) void frame_attn_kernel(
    const bf16_t* __restrict__ qh, const bf16_t* __restrict__ kh,
    const bf16_t* __restrict__ vh, bf16_t* __restrict__ attnout) {
  extern __shared__ char smem_raw[];
  bf16_t* Ksh = (bf16_t*)smem_raw;            // NKPAD * KST
  bf16_t* Vts = Ksh + NKPAD * KST;            // DIM_HEAD * VTST  (V transposed)
  bf16_t* Psh = Vts + DIM_HEAD * VTST;        // 8 waves * 16 * PST

  const int f  = blockIdx.x;
  const int bh = blockIdx.y;
  const int b = bh >> 3, h = bh & 7;
  const long base = (long)bh * SEQ;
  const int tid = threadIdx.x;

  // Stage K row-major (zero-pad keys >= 197)
  for (int idx = tid; idx < NKPAD * DIM_HEAD; idx += 256) {
    int kk = idx >> 6;
    int dd = idx & 63;
    bf16_t kv = (bf16_t)0.f;
    if (kk <= NPATCH) {
      long row = (kk == 0) ? base : (base + (long)f * NPATCH + kk);
      kv = kh[row * DIM_HEAD + dd];
    }
    Ksh[kk * KST + dd] = kv;
  }
  // Stage V transposed: Vt[dd][kk]; iterate kk-minor so LDS stores coalesce
  for (int idx = tid; idx < DIM_HEAD * 256; idx += 256) {
    int dd = idx >> 8;
    int kk = idx & 255;
    if (kk < NKPAD) {
      bf16_t vv = (bf16_t)0.f;
      if (kk <= NPATCH) {
        long row = (kk == 0) ? base : (base + (long)f * NPATCH + kk);
        vv = vh[row * DIM_HEAD + dd];
      }
      Vts[dd * VTST + kk] = vv;
    }
  }
  __syncthreads();

  const int lane = tid & 31;
  const int wave = tid >> 5;
  const int am  = lane & 15;
  const int kb  = (lane >> 4) * 8;
  const int grp = (lane >> 4) << 3;           // C-layout row group: 0 or 8
  bf16_t* pw = Psh + wave * 16 * PST;

  for (int t = wave; t < 13; t += 8) {        // 13 query tiles of 16 (196 rows)
    const int qbase = t * 16;
    int qrow = qbase + am; if (qrow > NPATCH - 1) qrow = NPATCH - 1;
    const bf16_t* qp = qh + (base + 1 + (long)f * NPATCH + qrow) * DIM_HEAD;
    v16bf aq0 = ld_frag(qp + kb);
    v16bf aq1 = ld_frag(qp + 32 + kb);

    // S = Q K^T  (14 key tiles kept in registers, f32)
    float S[14][8];
#pragma unroll
    for (int jt = 0; jt < 14; ++jt) {
      const bf16_t* kp = Ksh + (jt * 16 + am) * KST;
      v16bf bk0 = ld_frag(kp + kb);
      v16bf bk1 = ld_frag(kp + 32 + kb);
      v8f c = {};
      c = wmma_bf16(aq0, bk0, c);
      c = wmma_bf16(aq1, bk1, c);
#pragma unroll
      for (int r = 0; r < 8; ++r) S[jt][r] = c[r];
    }

    // Row softmax: reductions across the 16 lanes of each half-wave group
    float rinv[8];
#pragma unroll
    for (int r = 0; r < 8; ++r) {
      float mx = -1e30f;
#pragma unroll
      for (int jt = 0; jt < 14; ++jt) {
        int key = jt * 16 + am;
        if (key >= NPATCH + 1) S[jt][r] = -1e30f;   // mask padded keys
        mx = fmaxf(mx, S[jt][r]);
      }
#pragma unroll
      for (int off = 1; off < 16; off <<= 1) mx = fmaxf(mx, __shfl_xor(mx, off, 16));
      float sm = 0.f;
#pragma unroll
      for (int jt = 0; jt < 14; ++jt) {
        float p = __expf(S[jt][r] - mx);
        S[jt][r] = p;
        sm += p;
      }
#pragma unroll
      for (int off = 1; off < 16; off <<= 1) sm += __shfl_xor(sm, off, 16);
      rinv[r] = 1.f / sm;
      const int prow = r + grp;
#pragma unroll
      for (int jt = 0; jt < 14; ++jt)
        pw[prow * PST + jt * 16 + am] = (bf16_t)S[jt][r];   // C-layout -> LDS
    }
    asm volatile("s_wait_dscnt 0" ::: "memory");  // same-wave LDS RAW fence

    // O = P V   (contract over 224 keys = 7 WMMAs of K=32, per 16-col d tile)
#pragma unroll
    for (int dt = 0; dt < 4; ++dt) {
      const bf16_t* vtr = Vts + (dt * 16 + am) * VTST;  // B col = d, contig keys
      v8f o = {};
#pragma unroll
      for (int kt = 0; kt < 7; ++kt) {
        const int k0 = kt * 32;
        v16bf ap = ld_frag(pw + am * PST + k0 + kb);
        v16bf bv = ld_frag(vtr + k0 + kb);
        o = wmma_bf16(ap, bv, o);
      }
#pragma unroll
      for (int r = 0; r < 8; ++r) {
        int q = qbase + r + grp;
        if (q < NPATCH) {
          float val = o[r] * rinv[r];
          attnout[((long)b * SEQ + 1 + (long)f * NPATCH + q) * DIM +
                  h * DIM_HEAD + dt * 16 + am] = (bf16_t)val;
        }
      }
    }
  }
}

// ---------------------------------------------------------------- output proj
// wt transposed: wt[n][k].
__global__ __launch_bounds__(256) void out_proj_kernel(
    const bf16_t* __restrict__ a, const bf16_t* __restrict__ wt,
    const float* __restrict__ bias, float* __restrict__ out) {
  const int lane = threadIdx.x & 31;
  const int wave = threadIdx.x >> 5;
  const int mt = blockIdx.y * 8 + wave;
  if (mt >= MTILES) return;
  const int m0 = mt * 16;
  const int n0 = blockIdx.x * 16;
  const int am = lane & 15;
  const int kb = (lane >> 4) * 8;
  const int grp = (lane >> 4) << 3;
  int arow = m0 + am; if (arow > MROWS - 1) arow = MROWS - 1;
  const bf16_t* ar = a + (long)arow * DIM;
  const bf16_t* wr = wt + (long)(n0 + am) * DIM;

  v8f acc = {};
  for (int k0 = 0; k0 < DIM; k0 += 32) {
    __builtin_prefetch(ar + k0 + 128, 0, 0);
    v16bf av = ld_frag(ar + k0 + kb);
    v16bf bm = ld_frag(wr + k0 + kb);
    acc = wmma_bf16(av, bm, acc);
  }
  const int col = n0 + am;
  const float bb = bias[col];
#pragma unroll
  for (int r = 0; r < 8; ++r) {
    int m = m0 + r + grp;
    if (m < MROWS) out[(long)m * DIM + col] = acc[r] + bb;
  }
}

// ---------------------------------------------------------------- launch
extern "C" void kernel_launch(void* const* d_in, const int* in_sizes, int n_in,
                              void* d_out, int out_size, void* d_ws, size_t ws_size,
                              hipStream_t stream) {
  const float* x    = (const float*)d_in[0];
  const float* Wqkv = (const float*)d_in[1];
  const float* Wout = (const float*)d_in[2];
  const float* bout = (const float*)d_in[3];
  // d_in[4] is f == 16, baked into FRAMES.
  float* out = (float*)d_out;

  char* ws = (char*)d_ws;
  size_t off = 0;
  bf16_t* wqkvt   = (bf16_t*)(ws + off); off += (size_t)DIM * NQKV * sizeof(bf16_t);
  bf16_t* woutt   = (bf16_t*)(ws + off); off += (size_t)DIM * DIM * sizeof(bf16_t);
  bf16_t* xh      = (bf16_t*)(ws + off); off += (size_t)MROWS * DIM * sizeof(bf16_t);
  bf16_t* qh      = (bf16_t*)(ws + off); off += (size_t)BATCH * HEADS * SEQ * DIM_HEAD * sizeof(bf16_t);
  bf16_t* kh      = (bf16_t*)(ws + off); off += (size_t)BATCH * HEADS * SEQ * DIM_HEAD * sizeof(bf16_t);
  bf16_t* vh      = (bf16_t*)(ws + off); off += (size_t)BATCH * HEADS * SEQ * DIM_HEAD * sizeof(bf16_t);
  bf16_t* attnout = (bf16_t*)(ws + off); off += (size_t)MROWS * DIM * sizeof(bf16_t);
  (void)ws_size; (void)in_sizes; (void)n_in; (void)out_size;

  convert_transpose_kernel<<<(DIM * NQKV + 255) / 256, 256, 0, stream>>>(Wqkv, wqkvt, DIM, NQKV);
  convert_transpose_kernel<<<(DIM * DIM + 255) / 256, 256, 0, stream>>>(Wout, woutt, DIM, DIM);
  convert_bf16_kernel<<<(MROWS * DIM + 255) / 256, 256, 0, stream>>>(x, xh, MROWS * DIM);

  qkv_gemm_kernel<<<dim3(NQKV / 16, (MTILES + 7) / 8), 256, 0, stream>>>(xh, wqkvt, qh, kh, vh);

  cls_attn_kernel<<<BATCH * HEADS, 256, 0, stream>>>(qh, kh, vh, attnout);

  size_t smem = (size_t)(NKPAD * KST + DIM_HEAD * VTST + 8 * 16 * PST) * sizeof(bf16_t);  // ~119 KB of the 320 KB WGP LDS
  frame_attn_kernel<<<dim3(FRAMES, BATCH * HEADS), 256, smem, stream>>>(qh, kh, vh, attnout);

  out_proj_kernel<<<dim3(DIM / 16, (MTILES + 7) / 8), 256, 0, stream>>>(attnout, woutt, bout, out);
}
